// PoseStreamLayer_62491774157539
// MI455X (gfx1250) — matrix operand order repaired and checked by reading, not compiled, not run on hardware
//
#include <hip/hip_runtime.h>
#include <hip/hip_bf16.h>

typedef __attribute__((ext_vector_type(16))) _Float16 v16h;
typedef __attribute__((ext_vector_type(8)))  float    v8f;

#define DT   128      // T
#define DKV  3        // K (views)
#define DP   2        // P
#define DJ   17       // J
#define DD   128      // D model
#define DH   8        // heads
#define DHD  16       // head dim
#define NTOK (DT*DKV*DP*DJ)   // 13056
#define WIN  16               // temporal half-window
#define NWIN (2*WIN+1)        // 33
#define NEGV (-1e30f)

// ---------------------------------------------------------------------------
// Fused GEMM: out[M,N] = act( A[M,KD] * W[N,KD]^T + bias[N] ) (+ resid[M,N])
// One wave32 computes one 16x16 tile via v_wmma_f32_16x16x32_f16.
// KD templated so the K-loop fully unrolls (4 or 8 WMMAs in-line).
// ---------------------------------------------------------------------------
template<int KD>
__global__ __launch_bounds__(256) void gemm_wmma_kernel(
    const float* __restrict__ A, const float* __restrict__ W,
    const float* __restrict__ bias, const float* __restrict__ resid,
    float* __restrict__ out, int M, int N, int act)
{
  const int lane   = threadIdx.x & 31;
  const int wave   = threadIdx.x >> 5;
  const int tilesN = N >> 4;
  const int tile   = blockIdx.x * 8 + wave;
  if (tile >= (M >> 4) * tilesN) return;
  const int tm = tile / tilesN;
  const int tn = tile - tm * tilesN;

  const int mr = lane & 15;     // row (A) / col (B) within tile
  const int g  = lane >> 4;     // lane half
  const float* __restrict__ Arow = A + (long)(tm * 16 + mr) * KD;
  const float* __restrict__ Wrow = W + (long)(tn * 16 + mr) * KD;
  __builtin_prefetch(Wrow, 0, 3);   // global_prefetch_b8 for the weight row

  v8f acc = {};
#pragma unroll
  for (int kk = 0; kk < KD; kk += 32) {
    v16h af, bf;
#pragma unroll
    for (int e = 0; e < 16; ++e) {
      const int v  = e >> 1, b0 = e & 1;
      // A 16x32 f16 layout: K = 16*(v/4) + 8*g + 2*(v%4) + b
      const int ka = kk + ((v >> 2) << 4) + (g << 3) + ((v & 3) << 1) + b0;
      // B 32x16 f16 layout: K = 16*(v/4) + 8*g + 2*(v%4) + b
      const int kb = kk + ((v >> 2) << 4) + (g << 3) + ((v & 3) << 1) + b0;
      af[e] = (_Float16)Arow[ka];
      bf[e] = (_Float16)Wrow[kb];
    }
    acc = __builtin_amdgcn_wmma_f32_16x16x32_f16(false, af, false, bf,
                                                 (short)0, acc, false, false);
  }

  const int   col = tn * 16 + mr;
  const float bv  = bias[col];
#pragma unroll
  for (int v = 0; v < 8; ++v) {
    const int  row = tm * 16 + (g << 3) + v;   // C/D layout: M = v + 8*g
    float val = acc[v] + bv;
    if (act) val = fmaxf(val, 0.f);
    const long idx = (long)row * N + col;
    if (resid) val += resid[idx];
    out[idx] = val;
  }
}

// ---------------------------------------------------------------------------
// LayerNorm over D=128, one wave per output row; wave32 shuffle reductions.
// mode 0: identity row order; mode 1: gather into view-attn order (t,p,j,k);
// mode 2: gather into temporal order (k,p,j,t) + sinusoidal PE.
// ---------------------------------------------------------------------------
__global__ __launch_bounds__(256) void ln_kernel(
    const float* __restrict__ X, const float* __restrict__ gw,
    const float* __restrict__ bw, float* __restrict__ out, int mode)
{
  const int lane = threadIdx.x & 31;
  const int wave = threadIdx.x >> 5;
  const int r = blockIdx.x * 8 + wave;            // destination row
  if (r >= NTOK) return;

  int src = r, t = 0;
  if (mode == 1) {
    const int k = r % DKV; int q = r / DKV;
    const int j = q % DJ;  q /= DJ;
    const int p = q % DP;  const int tt = q / DP;
    src = ((tt * DKV + k) * DP + p) * DJ + j;
  } else if (mode == 2) {
    t = r & (DT - 1);
    int n = r >> 7;
    const int j = n % DJ; n /= DJ;
    const int p = n % DP; const int k = n / DP;
    src = ((t * DKV + k) * DP + p) * DJ + j;
  }

  const float* __restrict__ xr = X + (long)src * DD;
  const float x0 = xr[lane], x1 = xr[lane + 32], x2 = xr[lane + 64], x3 = xr[lane + 96];
  float s = x0 + x1 + x2 + x3;
#pragma unroll
  for (int off = 16; off; off >>= 1) s += __shfl_xor(s, off, 32);
  const float mean = s * (1.f / DD);
  const float d0 = x0 - mean, d1 = x1 - mean, d2 = x2 - mean, d3 = x3 - mean;
  float v = d0 * d0 + d1 * d1 + d2 * d2 + d3 * d3;
#pragma unroll
  for (int off = 16; off; off >>= 1) v += __shfl_xor(v, off, 32);
  const float inv = rsqrtf(v * (1.f / DD) + 1e-5f);

  float* __restrict__ orow = out + (long)r * DD;
#pragma unroll
  for (int i = 0; i < 4; ++i) {
    const int d = lane + 32 * i;
    const float dv = (i == 0) ? d0 : (i == 1) ? d1 : (i == 2) ? d2 : d3;
    float val = dv * inv * gw[d] + bw[d];
    if (mode == 2) {
      const float div = __expf(-(float)(d & ~1) * (9.2103403719761836f / DD));
      const float ang = (float)t * div;
      val += (d & 1) ? __cosf(ang) : __sinf(ang);
    }
    orow[d] = val;
  }
}

// ---------------------------------------------------------------------------
// Joint attention: per (seq n, head h); lane = query row (J=17), dh=16.
// qkv rows: [q(0..127) | k(128..255) | v(256..383)] per token.
// ---------------------------------------------------------------------------
__global__ __launch_bounds__(32) void attn_joint_kernel(
    const float* __restrict__ qkv, const float* __restrict__ mask,
    float* __restrict__ out)
{
  const int n = blockIdx.x >> 3;
  const int h = blockIdx.x & 7;
  const int l = threadIdx.x;
  if (l >= DJ) return;

  float q[DHD];
  const float* qr = qkv + (long)(n * DJ + l) * 384 + h * DHD;
#pragma unroll
  for (int d = 0; d < DHD; ++d) q[d] = qr[d];

  float s[DJ]; float mx = NEGV;
  const float* mrow = mask + ((long)(n * DH + h) * DJ + l) * DJ;
#pragma unroll
  for (int c = 0; c < DJ; ++c) {
    const float* kr = qkv + (long)(n * DJ + c) * 384 + DD + h * DHD;
    float dot = 0.f;
#pragma unroll
    for (int d = 0; d < DHD; ++d) dot += q[d] * kr[d];
    s[c] = dot * 0.25f + mrow[c];
    mx = fmaxf(mx, s[c]);
  }
  float sum = 0.f;
#pragma unroll
  for (int c = 0; c < DJ; ++c) { const float e = __expf(s[c] - mx); s[c] = e; sum += e; }
  const float inv = 1.f / sum;

  float o[DHD] = {};
#pragma unroll
  for (int c = 0; c < DJ; ++c) {
    const float w = s[c] * inv;
    const float* vr = qkv + (long)(n * DJ + c) * 384 + 2 * DD + h * DHD;
#pragma unroll
    for (int d = 0; d < DHD; ++d) o[d] += w * vr[d];
  }
  float* orow = out + (long)(n * DJ + l) * DD + h * DHD;
#pragma unroll
  for (int d = 0; d < DHD; ++d) orow[d] = o[d];
}

// ---------------------------------------------------------------------------
// View attention: K=3; one wave handles all 8 heads x 3 rows of one sequence.
// ---------------------------------------------------------------------------
__global__ __launch_bounds__(32) void attn_view_kernel(
    const float* __restrict__ qkv, const float* __restrict__ mask,
    float* __restrict__ out)
{
  const int n = blockIdx.x;                       // 0..4351
  const int l = threadIdx.x;
  if (l >= DH * DKV) return;                      // 24 active lanes
  const int h = l / DKV, r = l % DKV;

  float q[DHD];
  const float* qr = qkv + (long)(n * DKV + r) * 384 + h * DHD;
#pragma unroll
  for (int d = 0; d < DHD; ++d) q[d] = qr[d];

  float s[DKV]; float mx = NEGV;
  const float* mrow = mask + ((long)(n * DH + h) * DKV + r) * DKV;
#pragma unroll
  for (int c = 0; c < DKV; ++c) {
    const float* kr = qkv + (long)(n * DKV + c) * 384 + DD + h * DHD;
    float dot = 0.f;
#pragma unroll
    for (int d = 0; d < DHD; ++d) dot += q[d] * kr[d];
    s[c] = dot * 0.25f + mrow[c];
    mx = fmaxf(mx, s[c]);
  }
  float sum = 0.f;
#pragma unroll
  for (int c = 0; c < DKV; ++c) { const float e = __expf(s[c] - mx); s[c] = e; sum += e; }
  const float inv = 1.f / sum;

  float o[DHD] = {};
#pragma unroll
  for (int c = 0; c < DKV; ++c) {
    const float w = s[c] * inv;
    const float* vr = qkv + (long)(n * DKV + c) * 384 + 2 * DD + h * DHD;
#pragma unroll
    for (int d = 0; d < DHD; ++d) o[d] += w * vr[d];
  }
  float* orow = out + (long)(n * DKV + r) * DD + h * DHD;
#pragma unroll
  for (int d = 0; d < DHD; ++d) orow[d] = o[d];
}

// ---------------------------------------------------------------------------
// Windowed temporal attention: block = (seq n, head h), thread = query t.
// Window of 33 with boundary masking + temporal cost-mask add.
// ---------------------------------------------------------------------------
__global__ __launch_bounds__(128) void attn_temporal_kernel(
    const float* __restrict__ qkv, const float* __restrict__ cm,
    float* __restrict__ out)
{
  const int n = blockIdx.x >> 3;
  const int h = blockIdx.x & 7;
  const int t = threadIdx.x;

  float q[DHD];
  const float* qr = qkv + (long)(n * DT + t) * 384 + h * DHD;
#pragma unroll
  for (int d = 0; d < DHD; ++d) q[d] = qr[d];

  float s[NWIN]; float mx = NEGV;
  const float* cmr = cm + ((long)(n * DH + h) * DT + t) * DT;
  for (int w = 0; w < NWIN; ++w) {
    const int sp = t - WIN + w;
    float sc = NEGV;
    if (sp >= 0 && sp < DT) {
      const float* kr = qkv + (long)(n * DT + sp) * 384 + DD + h * DHD;
      float dot = 0.f;
#pragma unroll
      for (int d = 0; d < DHD; ++d) dot += q[d] * kr[d];
      sc = dot * 0.25f + cmr[sp];
    }
    s[w] = sc;
    mx = fmaxf(mx, sc);
  }
  float sum = 0.f;
  for (int w = 0; w < NWIN; ++w) { const float e = __expf(s[w] - mx); s[w] = e; sum += e; }
  const float inv = 1.f / sum;

  float o[DHD] = {};
  for (int w = 0; w < NWIN; ++w) {
    const int sp = t - WIN + w;
    if (sp < 0 || sp >= DT) continue;
    const float wg = s[w] * inv;
    const float* vr = qkv + (long)(n * DT + sp) * 384 + 2 * DD + h * DHD;
#pragma unroll
    for (int d = 0; d < DHD; ++d) o[d] += wg * vr[d];
  }
  float* orow = out + (long)(n * DT + t) * DD + h * DHD;
#pragma unroll
  for (int d = 0; d < DHD; ++d) orow[d] = o[d];
}

// ---------------------------------------------------------------------------
// Residual add with permuted gather: out[tok] = base[tok] + delta[perm(tok)]
// mode 1: delta rows in view order; mode 2: delta rows in temporal order.
// ---------------------------------------------------------------------------
__global__ __launch_bounds__(256) void add_perm_kernel(
    const float* __restrict__ base, const float* __restrict__ delta,
    float* __restrict__ out, int mode)
{
  const long gid = (long)blockIdx.x * 256 + threadIdx.x;
  if (gid >= (long)NTOK * DD) return;
  const int row = (int)(gid >> 7);
  const int d   = (int)(gid & 127);
  const int j = row % DJ; int q = row / DJ;
  const int p = q % DP;   q /= DP;
  const int k = q % DKV;  const int t = q / DKV;
  long prow;
  if (mode == 1) prow = (long)((t * DP + p) * DJ + j) * DKV + k;
  else           prow = (long)((k * DP + p) * DJ + j) * DT  + t;
  out[gid] = base[gid] + delta[prow * DD + d];
}

// ---------------------------------------------------------------------------
extern "C" void kernel_launch(void* const* d_in, const int* in_sizes, int n_in,
                              void* d_out, int out_size, void* d_ws, size_t ws_size,
                              hipStream_t stream)
{
  (void)in_sizes; (void)n_in; (void)out_size; (void)ws_size;
  const float* x             = (const float*)d_in[0];
  const float* joint_mask    = (const float*)d_in[1];
  const float* view_mask     = (const float*)d_in[2];
  const float* temporal_mask = (const float*)d_in[3];
  const float* jn_g  = (const float*)d_in[4];
  const float* jn_b  = (const float*)d_in[5];
  const float* j_wqkv= (const float*)d_in[6];
  const float* j_bqkv= (const float*)d_in[7];
  const float* j_wo  = (const float*)d_in[8];
  const float* j_bo  = (const float*)d_in[9];
  const float* vn_g  = (const float*)d_in[10];
  const float* vn_b  = (const float*)d_in[11];
  const float* v_wqkv= (const float*)d_in[12];
  const float* v_bqkv= (const float*)d_in[13];
  const float* v_wo  = (const float*)d_in[14];
  const float* v_bo  = (const float*)d_in[15];
  const float* tn_g  = (const float*)d_in[16];
  const float* tn_b  = (const float*)d_in[17];
  const float* t_wqkv= (const float*)d_in[18];
  const float* t_bqkv= (const float*)d_in[19];
  const float* t_wo  = (const float*)d_in[20];
  const float* t_bo  = (const float*)d_in[21];
  const float* fn_g  = (const float*)d_in[22];
  const float* fn_b  = (const float*)d_in[23];
  const float* ff_w1 = (const float*)d_in[24];
  const float* ff_b1 = (const float*)d_in[25];
  const float* ff_w2 = (const float*)d_in[26];
  const float* ff_b2 = (const float*)d_in[27];
  float* out = (float*)d_out;

  float* ws       = (float*)d_ws;
  float* buf_h    = ws;                                // NTOK*128
  float* buf_qkv  = buf_h    + (long)NTOK * DD;        // NTOK*384
  float* buf_attn = buf_qkv  + (long)NTOK * 384;       // NTOK*128
  float* buf_x1   = buf_attn + (long)NTOK * DD;        // NTOK*128
  float* buf_x2   = buf_x1   + (long)NTOK * DD;        // NTOK*128

  const int lnGrid   = NTOK / 8;                       // 1632
  const int g384     = (NTOK / 16) * (384 / 16) / 8;   // 2448
  const int g128     = (NTOK / 16) * (128 / 16) / 8;   // 816
  const int g256     = (NTOK / 16) * (256 / 16) / 8;   // 1632
  const int ewGrid   = (NTOK * DD) / 256;              // 6528

  // ---- stage 1: joint attention ----
  ln_kernel<<<lnGrid, 256, 0, stream>>>(x, jn_g, jn_b, buf_h, 0);
  gemm_wmma_kernel<128><<<g384, 256, 0, stream>>>(buf_h, j_wqkv, j_bqkv, nullptr,
                                                  buf_qkv, NTOK, 384, 0);
  attn_joint_kernel<<<(NTOK / DJ) * DH, 32, 0, stream>>>(buf_qkv, joint_mask, buf_attn);
  gemm_wmma_kernel<128><<<g128, 256, 0, stream>>>(buf_attn, j_wo, j_bo, x,
                                                  buf_x1, NTOK, 128, 0);

  // ---- stage 2: view attention ----
  ln_kernel<<<lnGrid, 256, 0, stream>>>(buf_x1, vn_g, vn_b, buf_h, 1);
  gemm_wmma_kernel<128><<<g384, 256, 0, stream>>>(buf_h, v_wqkv, v_bqkv, nullptr,
                                                  buf_qkv, NTOK, 384, 0);
  attn_view_kernel<<<NTOK / DKV, 32, 0, stream>>>(buf_qkv, view_mask, buf_attn);
  gemm_wmma_kernel<128><<<g128, 256, 0, stream>>>(buf_attn, v_wo, v_bo, nullptr,
                                                  buf_h, NTOK, 128, 0);
  add_perm_kernel<<<ewGrid, 256, 0, stream>>>(buf_x1, buf_h, buf_x2, 1);

  // ---- stage 3: windowed temporal attention ----
  ln_kernel<<<lnGrid, 256, 0, stream>>>(buf_x2, tn_g, tn_b, buf_h, 2);
  gemm_wmma_kernel<128><<<g384, 256, 0, stream>>>(buf_h, t_wqkv, t_bqkv, nullptr,
                                                  buf_qkv, NTOK, 384, 0);
  attn_temporal_kernel<<<(NTOK / (DJ * DP * DKV)) * DH, 128, 0, stream>>>(
      buf_qkv, temporal_mask, buf_attn);
  gemm_wmma_kernel<128><<<g128, 256, 0, stream>>>(buf_attn, t_wo, t_bo, nullptr,
                                                  buf_h, NTOK, 128, 0);
  add_perm_kernel<<<ewGrid, 256, 0, stream>>>(buf_x2, buf_h, buf_x1, 2);

  // ---- stage 4: FFN ----
  ln_kernel<<<lnGrid, 256, 0, stream>>>(buf_x1, fn_g, fn_b, buf_h, 0);
  gemm_wmma_kernel<128><<<g256, 256, 0, stream>>>(buf_h, ff_w1, ff_b1, nullptr,
                                                  buf_qkv, NTOK, 256, 1 /*relu*/);
  gemm_wmma_kernel<256><<<g128, 256, 0, stream>>>(buf_qkv, ff_w2, ff_b2, buf_x1,
                                                  out, NTOK, 128, 0);
}
